// IntraAttention_13829794693130
// MI455X (gfx1250) — matrix operand order repaired and checked by reading, not compiled, not run on hardware
//
#include <hip/hip_runtime.h>

// ---------------------------------------------------------------------------
// IntraAttention for MI455X (gfx1250, wave32, WMMA bf16).
//   xb = bf16(x); WT = bf16(W^T)                (elementwise passes)
//   f  = x@W + b          (bf16 WMMA, f32 accum; 32x32 tile per wave)
//   e  = f f^T + dist     (bf16 WMMA, f32 accum, flash-style online softmax)
//   out = softmax(e) f    (bf16 WMMA, f32 accum)
// ---------------------------------------------------------------------------

typedef __attribute__((ext_vector_type(16))) __bf16 v16bf;
typedef __attribute__((ext_vector_type(8)))  float  v8f;

#define B_  8
#define S_  2048
#define D_  1024

union V16U { uint4 u[2]; v16bf v; };

static __device__ inline unsigned bf16pack2(float lo, float hi) {
    union { __bf16 h; unsigned short u; } a, b;
    a.h = (__bf16)lo; b.h = (__bf16)hi;
    return ((unsigned)b.u << 16) | (unsigned)a.u;
}

// ---------------------------------------------------------------------------
// Kernel 0a: W [K][N] fp32  ->  WT [N][K] bf16  (contiguous B-frag loads)
// ---------------------------------------------------------------------------
__global__ __launch_bounds__(256) void k_convert_wt(const float* __restrict__ W,
                                                    __bf16* __restrict__ WT) {
    int idx = blockIdx.x * 256 + threadIdx.x;       // 0 .. 1M-1
    int n = idx >> 10, k = idx & 1023;
    WT[(size_t)n * D_ + k] = (__bf16)W[(size_t)k * D_ + n];
}

// ---------------------------------------------------------------------------
// Kernel 0b: x fp32 -> xb bf16 (once; removes per-N-block re-conversion and
// halves A-side bytes in the projection GEMM)
// ---------------------------------------------------------------------------
__global__ __launch_bounds__(256) void k_convert_x(const float* __restrict__ x,
                                                   __bf16* __restrict__ xb) {
    size_t i = ((size_t)blockIdx.x * 256 + threadIdx.x) * 4;
    float4 v = *(const float4*)(x + i);
    union { __bf16 h[4]; uint2 u; } o;
    o.h[0] = (__bf16)v.x; o.h[1] = (__bf16)v.y;
    o.h[2] = (__bf16)v.z; o.h[3] = (__bf16)v.w;
    *(uint2*)(xb + i) = o.u;
}

// ---------------------------------------------------------------------------
// Kernel 1: f = x@W + b.  Block = 32 rows x 256 cols, 8 waves.
// Wave = 32x32 tile (2 M-tiles x 2 N-tiles): 4 wmma per 8 b128 loads per K-step.
// Writes f [B*S][D] bf16 and fT [B][D][S] bf16.
// ---------------------------------------------------------------------------
__global__ __launch_bounds__(256) void k_gemm_f(const __bf16* __restrict__ xb,
                                                const __bf16* __restrict__ WT,
                                                const float*  __restrict__ bias,
                                                __bf16* __restrict__ f,
                                                __bf16* __restrict__ fT) {
    const int lane = threadIdx.x & 31;
    const int wave = threadIdx.x >> 5;
    const int lm   = lane & 15;          // A row / B col / C col within tile
    const int hi   = lane >> 4;          // lane half: K sub-range / C row+8
    const int m0   = blockIdx.x * 32;
    const int n0   = blockIdx.y * 256 + wave * 32;

    v8f acc[2][2];
    v8f zero = {};
    acc[0][0] = zero; acc[0][1] = zero; acc[1][0] = zero; acc[1][1] = zero;

    const __bf16* arow0 = xb + (size_t)(m0 + lm)      * D_;
    const __bf16* arow1 = xb + (size_t)(m0 + 16 + lm) * D_;
    const __bf16* brow0 = WT + (size_t)(n0 + lm)      * D_;
    const __bf16* brow1 = WT + (size_t)(n0 + 16 + lm) * D_;

    for (int k0 = 0; k0 < D_; k0 += 32) {
        // A frags: K runs [8hi,8hi+8) and [16+8hi,16+8hi+8)
        V16U a0, a1;
        a0.u[0] = *(const uint4*)(arow0 + k0 + 8 * hi);
        a0.u[1] = *(const uint4*)(arow0 + k0 + 8 * hi + 16);
        a1.u[0] = *(const uint4*)(arow1 + k0 + 8 * hi);
        a1.u[1] = *(const uint4*)(arow1 + k0 + 8 * hi + 16);
        // B frags: K runs [8hi,8hi+8) and [16+8hi,16+8hi+8)
        V16U b0, b1;
        b0.u[0] = *(const uint4*)(brow0 + k0 + 8 * hi);
        b0.u[1] = *(const uint4*)(brow0 + k0 + 8 * hi + 16);
        b1.u[0] = *(const uint4*)(brow1 + k0 + 8 * hi);
        b1.u[1] = *(const uint4*)(brow1 + k0 + 8 * hi + 16);

        acc[0][0] = __builtin_amdgcn_wmma_f32_16x16x32_bf16(false, a0.v, false, b0.v,
                                                            (short)0, acc[0][0], false, false);
        acc[0][1] = __builtin_amdgcn_wmma_f32_16x16x32_bf16(false, a0.v, false, b1.v,
                                                            (short)0, acc[0][1], false, false);
        acc[1][0] = __builtin_amdgcn_wmma_f32_16x16x32_bf16(false, a1.v, false, b0.v,
                                                            (short)0, acc[1][0], false, false);
        acc[1][1] = __builtin_amdgcn_wmma_f32_16x16x32_bf16(false, a1.v, false, b1.v,
                                                            (short)0, acc[1][1], false, false);
    }

    #pragma unroll
    for (int nt = 0; nt < 2; ++nt) {
        const int ncol = n0 + nt * 16 + lm;
        const float bval = bias[ncol];
        #pragma unroll
        for (int mt = 0; mt < 2; ++mt) {
            #pragma unroll
            for (int r = 0; r < 8; ++r) {
                int m = m0 + mt * 16 + r + 8 * hi;       // C row = r + 8*hi
                __bf16 h = (__bf16)(acc[mt][nt][r] + bval);
                f[(size_t)m * D_ + ncol] = h;
                int bidx = m >> 11, srow = m & (S_ - 1);
                fT[((size_t)bidx * D_ + ncol) * S_ + srow] = h;
            }
        }
    }
}

// ---------------------------------------------------------------------------
// Kernel 2: flash attention.  Block = (16 queries, 1 batch), 8 waves.
// Wave w owns D-chunk [128w,128w+128) for BOTH the score K-split and the output.
// ---------------------------------------------------------------------------
__global__ __launch_bounds__(256) void k_attn(const __bf16* __restrict__ f,
                                              const __bf16* __restrict__ fT,
                                              float* __restrict__ out) {
    __shared__ float  Pf[32][16];       // raw score partials  [key][q]
    __shared__ __bf16 Pb[16][32];       // normalized probs    [q][key]
    __shared__ float  m_s[16], l_s[16], alpha_s[16];

    const int lane = threadIdx.x & 31;
    const int wave = threadIdx.x >> 5;
    const int lm   = lane & 15;
    const int hi   = lane >> 4;
    const int i0   = blockIdx.x * 16;
    const int b    = blockIdx.y;
    const int c0   = wave * 128;

    const __bf16* fb  = f  + (size_t)b * S_ * D_;
    const __bf16* fTb = fT + (size_t)b * D_ * S_;

    if (threadIdx.x < 16) { m_s[threadIdx.x] = -1e30f; l_s[threadIdx.x] = 0.f; alpha_s[threadIdx.x] = 0.f; }

    // Preload score B-frags: B[k][n] = f[i0+n][k] over this wave's K-chunk.
    v16bf fBq[4];
    {
        const __bf16* qrow = fb + (size_t)(i0 + lm) * D_ + c0;
        #pragma unroll
        for (int st = 0; st < 4; ++st) {
            V16U t;
            const uint4* p = (const uint4*)(qrow + st * 32 + 8 * hi);
            t.u[0] = p[0]; t.u[1] = p[2];
            fBq[st] = t.v;
        }
    }

    v8f zero = {};
    v8f acc[8];
    #pragma unroll
    for (int t = 0; t < 8; ++t) acc[t] = zero;

    __syncthreads();

    for (int j0 = 0; j0 < S_; j0 += 32) {
        // ---- prefetch next key block (keys rows + value columns) ----------
        if (j0 + 32 < S_) {
            __builtin_prefetch(fb  + (size_t)(j0 + 32 + lm) * D_ + c0, 0, 1);
            __builtin_prefetch(fb  + (size_t)(j0 + 48 + lm) * D_ + c0, 0, 1);
            __builtin_prefetch(fTb + (size_t)(c0 + 8 * lane) * S_ + j0 + 32, 0, 1);
        }

        // ---- zero score buffer --------------------------------------------
        ((float*)Pf)[threadIdx.x]       = 0.f;
        ((float*)Pf)[threadIdx.x + 256] = 0.f;
        __syncthreads();

        // ---- scores (transposed): C[32 keys][16 q] over this wave's K-chunk
        v8f cs0 = zero, cs1 = zero;
        #pragma unroll
        for (int st = 0; st < 4; ++st) {
            int kb = c0 + st * 32;
            V16U a0, a1;
            const __bf16* r0 = fb + (size_t)(j0 + lm)      * D_ + kb + 8 * hi;
            const __bf16* r1 = fb + (size_t)(j0 + 16 + lm) * D_ + kb + 8 * hi;
            a0.u[0] = *(const uint4*)(r0);  a0.u[1] = *(const uint4*)(r0 + 16);
            a1.u[0] = *(const uint4*)(r1);  a1.u[1] = *(const uint4*)(r1 + 16);
            cs0 = __builtin_amdgcn_wmma_f32_16x16x32_bf16(false, a0.v, false, fBq[st],
                                                          (short)0, cs0, false, false);
            cs1 = __builtin_amdgcn_wmma_f32_16x16x32_bf16(false, a1.v, false, fBq[st],
                                                          (short)0, cs1, false, false);
        }
        // reduce K-split partials across the 8 waves
        #pragma unroll
        for (int r = 0; r < 8; ++r) {
            atomicAdd(&Pf[r + 8 * hi][lm],      cs0[r]);
            atomicAdd(&Pf[16 + r + 8 * hi][lm], cs1[r]);
        }
        __syncthreads();

        // ---- wave 0: bias + online softmax stats + normalized P in bf16 ---
        if (wave == 0) {
            const int   q   = lm;
            const float i_f = (float)(i0 + q);
            float mold = m_s[q], lold = l_s[q];
            float s[16];
            float mloc = -1e30f;
            #pragma unroll
            for (int c = 0; c < 16; ++c) {
                int jj = j0 + 16 * hi + c;
                float d = fminf(fabsf(i_f - (float)jj), 10.0f);
                float sv = Pf[16 * hi + c][q] + d * 0.01f;
                s[c] = sv;
                mloc = fmaxf(mloc, sv);
            }
            float mblk = fmaxf(mloc, __shfl_xor(mloc, 16, 32));
            float mnew = fmaxf(mold, mblk);
            float al   = __expf(mold - mnew);
            float lsum = 0.f;
            unsigned pk[8];
            #pragma unroll
            for (int c = 0; c < 8; ++c) {
                float p0 = __expf(s[2 * c]     - mnew);
                float p1 = __expf(s[2 * c + 1] - mnew);
                lsum += p0 + p1;
                pk[c] = bf16pack2(p0, p1);
            }
            float lblk = lsum + __shfl_xor(lsum, 16, 32);
            float lnew = lold * al + lblk;
            if (hi == 0) { m_s[q] = mnew; l_s[q] = lnew; alpha_s[q] = al; }
            uint4* dst = (uint4*)&Pb[q][16 * hi];
            dst[0] = make_uint4(pk[0], pk[1], pk[2], pk[3]);
            dst[1] = make_uint4(pk[4], pk[5], pk[6], pk[7]);
        }
        __syncthreads();

        // ---- rescale accumulator, then acc += P @ V ------------------------
        float alr[8];
        #pragma unroll
        for (int r = 0; r < 8; ++r) alr[r] = alpha_s[r + 8 * hi];
        #pragma unroll
        for (int t = 0; t < 8; ++t)
            #pragma unroll
            for (int r = 0; r < 8; ++r) acc[t][r] *= alr[r];

        V16U pa;   // P as A-frag: row q = lane%16, key runs 8hi.. / 16+8hi..
        pa.u[0] = *(const uint4*)&Pb[lm][8 * hi];
        pa.u[1] = *(const uint4*)&Pb[lm][16 + 8 * hi];
        #pragma unroll
        for (int t = 0; t < 8; ++t) {
            // V B-frag from fT: col fixed per lane, key runs 8hi.. / 16+8hi..
            const __bf16* vcol = fTb + (size_t)(c0 + t * 16 + lm) * S_ + j0 + 8 * hi;
            V16U vb;
            vb.u[0] = *(const uint4*)(vcol);
            vb.u[1] = *(const uint4*)(vcol + 16);
            acc[t] = __builtin_amdgcn_wmma_f32_16x16x32_bf16(false, pa.v, false, vb.v,
                                                             (short)0, acc[t], false, false);
        }
    }

    // ---- normalize by l and store fp32 output -----------------------------
    float linv[8];
    #pragma unroll
    for (int r = 0; r < 8; ++r) linv[r] = 1.0f / l_s[r + 8 * hi];
    #pragma unroll
    for (int t = 0; t < 8; ++t) {
        #pragma unroll
        for (int r = 0; r < 8; ++r) {
            int m = i0 + r + 8 * hi;
            out[((size_t)b * S_ + m) * D_ + c0 + t * 16 + lm] = acc[t][r] * linv[r];
        }
    }
}

// ---------------------------------------------------------------------------
extern "C" void kernel_launch(void* const* d_in, const int* in_sizes, int n_in,
                              void* d_out, int out_size, void* d_ws, size_t ws_size,
                              hipStream_t stream) {
    const float* x    = (const float*)d_in[0];
    const float* W    = (const float*)d_in[1];
    const float* bias = (const float*)d_in[2];
    float* out = (float*)d_out;

    // Workspace: WT (2 MB) | f (32 MB) | fT (32 MB) | xb (32 MB)  -- all bf16
    __bf16* WT = (__bf16*)d_ws;
    __bf16* f  = WT + (size_t)D_ * D_;
    __bf16* fT = f  + (size_t)B_ * S_ * D_;
    __bf16* xb = fT + (size_t)B_ * S_ * D_;

    k_convert_wt<<<(D_ * D_) / 256, 256, 0, stream>>>(W, WT);
    k_convert_x<<<(B_ * S_ * D_) / (256 * 4), 256, 0, stream>>>(x, xb);
    k_gemm_f<<<dim3(B_ * S_ / 32, D_ / 256), 256, 0, stream>>>(xb, WT, bias, f, fT);
    k_attn<<<dim3(S_ / 16, B_), 256, 0, stream>>>(f, fT, out);
}
